// MaskedPolicy_89309549953663
// MI455X (gfx1250) — compile-verified, matrix operand order checked by reading, not run on hardware
//
#include <hip/hip_runtime.h>

#define B_SZ 8192
#define OBSD 512
#define NROB 8
#define NHEADS 5
#define HSIZE 64
#define HSUM 320
#define HID 1024
#define NEGV -1.0e9f

typedef __attribute__((ext_vector_type(16))) __bf16 v16bf;
typedef __attribute__((ext_vector_type(8)))  float  v8f;

__device__ __forceinline__ unsigned short f2bf(float f) {
  unsigned u = __float_as_uint(f);
  u += 0x7fffu + ((u >> 16) & 1u);
  return (unsigned short)(u >> 16);
}
__device__ __forceinline__ float bf2f(unsigned short h) {
  return __uint_as_float(((unsigned)h) << 16);
}

// ---------------------------------------------------------------- converts --
__global__ __launch_bounds__(256)
void cast_f32_bf16(const float* __restrict__ in, unsigned short* __restrict__ out, int n) {
  int i = blockIdx.x * 256 + threadIdx.x;
  if (i < n) out[i] = f2bf(in[i]);
}

// out[c][r] = (bf16) in[r][c]; all dims multiples of 32; z-batched
__global__ __launch_bounds__(256)
void transpose_f32_to_bf16(const float* __restrict__ in, unsigned short* __restrict__ out,
                           int R, int Ccols, long izS, long ozS) {
  __shared__ float tile[32][33];
  const float* inz = in + (long)blockIdx.z * izS;
  unsigned short* outz = out + (long)blockIdx.z * ozS;
  int tx = threadIdx.x & 31, ty = threadIdx.x >> 5;
  int c0 = blockIdx.x * 32, r0 = blockIdx.y * 32;
#pragma unroll
  for (int j = 0; j < 32; j += 8)
    tile[ty + j][tx] = inz[(size_t)(r0 + ty + j) * Ccols + c0 + tx];
  __syncthreads();
#pragma unroll
  for (int j = 0; j < 32; j += 8)
    outz[(size_t)(c0 + ty + j) * R + r0 + tx] = f2bf(tile[tx][ty + j]);
}

// -------------------------------------------------------------------- GEMM --
// C = act(A[M,K](bf16) @ Wt[Ncols,K](bf16)^T + bias), z-batched over Wt/bias/C.
// Double-buffered async global->LDS pipeline: prefetch tile k+1 under WMMAs of
// tile k; in-order async completion lets s_wait_asynccnt 4 retire only tile k.
#define BM 128
#define BN 128
#define BK 32
#define LDK 48   // padded K-stride in shorts: 96B/row -> 32B-aligned fragments

__global__ __launch_bounds__(256)
void gemm_bf16_async(const unsigned short* __restrict__ A,
                     const unsigned short* __restrict__ WtBase, long wZ,
                     const float* __restrict__ biasBase, int bZ,
                     void* __restrict__ Cbase, int ldc, int cZ, int outBF,
                     int M, int Ncols, int K, int do_relu)
{
  __shared__ __align__(32) unsigned short As[2][BM * LDK];
  __shared__ __align__(32) unsigned short Ws[2][BN * LDK];

  const unsigned short* Wt = WtBase + (long)blockIdx.z * wZ;
  const float* bias        = biasBase + (long)blockIdx.z * bZ;

  const int tid  = threadIdx.x;
  const int lane = tid & 31;
  const int wid  = tid >> 5;
  const int wm   = wid >> 1;   // 0..3 : 32-row strip
  const int wn   = wid & 1;    // 0..1 : 64-col strip
  const int m0   = blockIdx.y * BM;
  const int n0   = blockIdx.x * BN;

  const unsigned asBase = (unsigned)(uintptr_t)(&As[0][0]);
  const unsigned wsBase = (unsigned)(uintptr_t)(&Ws[0][0]);

  // issue 4 async 16B/lane copies staging tile at k0 into buffer buf
  auto stage = [&](int k0, int buf) {
    const unsigned aB = asBase + (unsigned)(buf * BM * LDK * 2);
    const unsigned wB = wsBase + (unsigned)(buf * BN * LDK * 2);
#pragma unroll
    for (int it = 0; it < 2; ++it) {
      int c   = tid + it * 256;     // 0..511 chunks of 8 bf16
      int row = c >> 2;
      int ko  = (c & 3) * 8;
      {
        int grow = m0 + row; if (grow >= M) grow = M - 1;
        unsigned long long ga =
            (unsigned long long)(uintptr_t)(A + (size_t)grow * K + k0 + ko);
        unsigned lo = aB + (unsigned)((row * LDK + ko) * 2);
        asm volatile("global_load_async_to_lds_b128 %0, %1, off"
                     :: "v"(lo), "v"(ga) : "memory");
      }
      {
        int gcol = n0 + row; if (gcol >= Ncols) gcol = Ncols - 1;
        unsigned long long ga =
            (unsigned long long)(uintptr_t)(Wt + (size_t)gcol * K + k0 + ko);
        unsigned lo = wB + (unsigned)((row * LDK + ko) * 2);
        asm volatile("global_load_async_to_lds_b128 %0, %1, off"
                     :: "v"(lo), "v"(ga) : "memory");
      }
    }
  };

  v8f zacc = {};
  v8f acc[2][4];
#pragma unroll
  for (int i = 0; i < 2; ++i)
#pragma unroll
    for (int j = 0; j < 4; ++j) acc[i][j] = zacc;

  const int rA = lane & 15;
  const int kb = (lane >> 4) * 16;
  const int hi = lane >> 4;

  const int nk = K / BK;
  stage(0, 0);

  for (int kt = 0; kt < nk; ++kt) {
    if (kt + 1 < nk) {
      stage((kt + 1) * BK, (kt + 1) & 1);
      // retire only tile kt's 4 copies; tile kt+1 stays in flight under WMMAs
      asm volatile("s_wait_asynccnt 0x4" ::: "memory");
    } else {
      asm volatile("s_wait_asynccnt 0x0" ::: "memory");
    }
    __syncthreads();

    const unsigned short* Asb = As[kt & 1];
    const unsigned short* Wsb = Ws[kt & 1];
    v16bf a0 = *(const v16bf*)(Asb + (wm * 32 + 0 * 16 + rA) * LDK + kb);
    v16bf a1 = *(const v16bf*)(Asb + (wm * 32 + 1 * 16 + rA) * LDK + kb);
#pragma unroll
    for (int nt = 0; nt < 4; ++nt) {
      v16bf bb = *(const v16bf*)(Wsb + (wn * 64 + nt * 16 + rA) * LDK + kb);
      acc[0][nt] = __builtin_amdgcn_wmma_f32_16x16x32_bf16(
          false, a0, false, bb, (short)0, acc[0][nt], false, false);
      acc[1][nt] = __builtin_amdgcn_wmma_f32_16x16x32_bf16(
          false, a1, false, bb, (short)0, acc[1][nt], false, false);
    }
    __syncthreads();   // protect buffer (kt&1) before it is re-staged at kt+2
  }

  // --- epilogue: bias + optional ReLU; bf16 or fp32 output ---
  unsigned short* C16 = (unsigned short*)Cbase + (long)blockIdx.z * cZ;
  float*          C32 = (float*)Cbase + (long)blockIdx.z * cZ;
#pragma unroll
  for (int mt = 0; mt < 2; ++mt)
#pragma unroll
    for (int nt = 0; nt < 4; ++nt) {
      int col = n0 + wn * 64 + nt * 16 + rA;
      if (col >= Ncols) continue;
      float bcol = bias[col];
#pragma unroll
      for (int r = 0; r < 8; ++r) {
        int row = m0 + wm * 32 + mt * 16 + hi * 8 + r;
        if (row < M) {
          float v = acc[mt][nt][r] + bcol;
          if (do_relu) v = fmaxf(v, 0.f);
          if (outBF) C16[(size_t)row * ldc + col] = f2bf(v);
          else       C32[(size_t)row * ldc + col] = v;
        }
      }
    }
}

// ----------------------------------------------------------------- tail ops --
__global__ __launch_bounds__(256)
void critic_gemv(const unsigned short* __restrict__ hc, const float* __restrict__ Wc2,
                 const float* __restrict__ bc2, float* __restrict__ out)
{
  int lane = threadIdx.x & 31, wid = threadIdx.x >> 5;
  int row = blockIdx.x * 8 + wid;
  float s = 0.f;
  for (int i = lane; i < HID / 2; i += 32)
    s += bf2f(hc[(size_t)row * (HID / 2) + i]) * Wc2[i];
#pragma unroll
  for (int o = 16; o > 0; o >>= 1) s += __shfl_xor(s, o, 32);
  if (lane == 0) out[2 * B_SZ + row] = s + bc2[0];
}

__global__ __launch_bounds__(256)
void policy_softmax(const float* __restrict__ logits,
                    const int* __restrict__ masks,
                    const int* __restrict__ actions,
                    float* __restrict__ out)
{
  __shared__ float sLp[8], sEnt[8];
  const int b = blockIdx.x;
  const int tid = threadIdx.x, lane = tid & 31, wid = tid >> 5;
  float lp_acc = 0.f, ent_acc = 0.f;

  for (int i = 0; i < 5; ++i) {
    int g  = wid * 5 + i;               // 0..39
    int n  = g / NHEADS, hd = g % NHEADS;
    long base = (long)b * (NROB * HSUM) + (long)n * HSUM + hd * HSIZE;

    float x0 = masks[base + lane]      ? logits[base + lane]      : NEGV;
    float x1 = masks[base + lane + 32] ? logits[base + lane + 32] : NEGV;

    float m = fmaxf(x0, x1);
#pragma unroll
    for (int o = 16; o > 0; o >>= 1) m = fmaxf(m, __shfl_xor(m, o, 32));
    float e = __expf(x0 - m) + __expf(x1 - m);
#pragma unroll
    for (int o = 16; o > 0; o >>= 1) e += __shfl_xor(e, o, 32);
    float lse = m + __logf(e);

    float lp0 = x0 - lse, lp1 = x1 - lse;
    float ent = -(__expf(lp0) * lp0 + __expf(lp1) * lp1);
#pragma unroll
    for (int o = 16; o > 0; o >>= 1) ent += __shfl_xor(ent, o, 32);

    int a = actions[b * (NROB * NHEADS) + g];
    float xsel = (a & 32) ? x1 : x0;
    float xa = __shfl(xsel, a & 31, 32);
    bool valid = ((unsigned)a) < HSIZE;
    lp_acc  += valid ? (xa - lse) : -1000.0f;
    ent_acc += valid ? ent : 0.0f;
  }
  if (lane == 0) { sLp[wid] = lp_acc; sEnt[wid] = ent_acc; }
  __syncthreads();
  if (tid == 0) {
    float l = 0.f, en = 0.f;
#pragma unroll
    for (int w = 0; w < 8; ++w) { l += sLp[w]; en += sEnt[w]; }
    out[b] = l;
    out[B_SZ + b] = en;
  }
}

// ---------------------------------------------------------------- launcher --
extern "C" void kernel_launch(void* const* d_in, const int* in_sizes, int n_in,
                              void* d_out, int out_size, void* d_ws, size_t ws_size,
                              hipStream_t stream) {
  const float* obs     = (const float*)d_in[0];
  const int*   actions = (const int*)d_in[1];
  const int*   masks   = (const int*)d_in[2];
  const float* W1      = (const float*)d_in[3];
  const float* b1      = (const float*)d_in[4];
  const float* W2      = (const float*)d_in[5];
  const float* b2      = (const float*)d_in[6];
  const float* headsW  = (const float*)d_in[7];
  const float* headsb  = (const float*)d_in[8];
  const float* Wc1     = (const float*)d_in[9];
  const float* bc1     = (const float*)d_in[10];
  const float* Wc2     = (const float*)d_in[11];
  const float* bc2     = (const float*)d_in[12];
  float* out = (float*)d_out;

  // workspace carve-up (bf16 = unsigned short)
  char* p = (char*)d_ws;
  unsigned short* obs_bf  = (unsigned short*)p; p += (size_t)B_SZ * OBSD * 2;      // 8.4MB
  unsigned short* W1t     = (unsigned short*)p; p += (size_t)OBSD * HID * 2;       // 1MB  [HID][OBSD]
  unsigned short* W2t     = (unsigned short*)p; p += (size_t)HID * HID * 2;        // 2MB  [HID][HID]
  unsigned short* headsWt = (unsigned short*)p; p += (size_t)NROB * HSUM * HID * 2;// 5.2MB [n][HSUM][HID]
  unsigned short* Wc1t    = (unsigned short*)p; p += (size_t)(HID / 2) * HID * 2;  // 1MB  [HID/2][HID]
  unsigned short* h1      = (unsigned short*)p; p += (size_t)B_SZ * HID * 2;       // 16.8MB
  unsigned short* h2      = (unsigned short*)p; p += (size_t)B_SZ * HID * 2;       // 16.8MB
  unsigned short* hc      = (unsigned short*)p; p += (size_t)B_SZ * (HID / 2) * 2; // 8.4MB
  float*          logits  = (float*)p;                                             // 83.9MB

  dim3 blk(256);

  // one-time precision/layout conversion (bf16, weights transposed to [N][K])
  cast_f32_bf16<<<(B_SZ * OBSD + 255) / 256, blk, 0, stream>>>(obs, obs_bf, B_SZ * OBSD);
  transpose_f32_to_bf16<<<dim3(HID / 32, OBSD / 32, 1), blk, 0, stream>>>(
      W1, W1t, OBSD, HID, 0, 0);
  transpose_f32_to_bf16<<<dim3(HID / 32, HID / 32, 1), blk, 0, stream>>>(
      W2, W2t, HID, HID, 0, 0);
  transpose_f32_to_bf16<<<dim3(HSUM / 32, HID / 32, NROB), blk, 0, stream>>>(
      headsW, headsWt, HID, HSUM, (long)HID * HSUM, (long)HSUM * HID);
  transpose_f32_to_bf16<<<dim3((HID / 2) / 32, HID / 32, 1), blk, 0, stream>>>(
      Wc1, Wc1t, HID, HID / 2, 0, 0);

  // h1 = relu(obs @ W1 + b1)            [bf16 out]
  gemm_bf16_async<<<dim3(HID / BN, B_SZ / BM, 1), blk, 0, stream>>>(
      obs_bf, W1t, 0, b1, 0, h1, HID, 0, 1, B_SZ, HID, OBSD, 1);
  // h2 = relu(h1 @ W2 + b2)             [bf16 out]
  gemm_bf16_async<<<dim3(HID / BN, B_SZ / BM, 1), blk, 0, stream>>>(
      h1, W2t, 0, b2, 0, h2, HID, 0, 1, B_SZ, HID, HID, 1);
  // logits[:, n*320:(n+1)*320] = h2 @ headsW[n] + headsb[n]   [fp32 out]
  gemm_bf16_async<<<dim3((HSUM + BN - 1) / BN, B_SZ / BM, NROB), blk, 0, stream>>>(
      h2, headsWt, (long)HSUM * HID, headsb, HSUM,
      logits, NROB * HSUM, HSUM, 0, B_SZ, HSUM, HID, 0);
  // hc = relu(h2 @ Wc1 + bc1)           [bf16 out]
  gemm_bf16_async<<<dim3((HID / 2) / BN, B_SZ / BM, 1), blk, 0, stream>>>(
      h2, Wc1t, 0, bc1, 0, hc, HID / 2, 0, 1, B_SZ, HID / 2, HID, 1);

  critic_gemv<<<B_SZ / 8, blk, 0, stream>>>(hc, Wc2, bc2, out);
  policy_softmax<<<B_SZ, blk, 0, stream>>>(logits, masks, actions, out);
}
